// MultiHeadAttention_5136780886308
// MI455X (gfx1250) — matrix operand order repaired and checked by reading, not compiled, not run on hardware
//
#include <hip/hip_runtime.h>
#include <hip/hip_bf16.h>

// ---------------------------------------------------------------------------
// MI455X (gfx1250) fused multi-head attention layer, round 3.
// All matmuls on v_wmma_f32_16x16x32_f16 (f16 operands, f32 accumulate).
// Round-3 changes vs round 2 (scheduling-focused):
//  * GEMM inner loop: all 6 fragments (2 A + 4 B) loaded before the 8 WMMAs,
//    so the load clause issues up front and waits stagger across the WMMAs
//  * unconditional speculative prefetch (no branch in the hot loop)
//  * attn QK^T: two key tiles per iteration with independent accumulators
//    (interleaved WMMA dependency chains)
//  * attn P@V: even/odd split accumulators to break the serial WMMA RAW chain
// wave32 throughout; 256-thread blocks = 8 waves.
// ---------------------------------------------------------------------------

typedef __attribute__((ext_vector_type(16))) _Float16 v16h;
typedef __attribute__((ext_vector_type(8)))  _Float16 v8h;
typedef __attribute__((ext_vector_type(4)))  _Float16 v4h;
typedef __attribute__((ext_vector_type(8)))  float    v8f;
typedef __attribute__((ext_vector_type(4)))  float    v4f;

#define D_MODEL 1024
#define SEQ     2048
#define BATCH   2
#define HEADS   16
#define DH      64
#define MTOK    (BATCH * SEQ)   /* 4096 token rows */

__device__ __forceinline__ int lane_id() { return (int)(threadIdx.x & 31); }

#define WMMA_F16(a, b, c) \
  __builtin_amdgcn_wmma_f32_16x16x32_f16(false, (a), false, (b), (short)0, (c), false, false)

// ---------------- WMMA fragment loaders (layouts per CDNA5 ISA) --
// A 16x32 f16: lane -> row (lane&15); lanes 0-15 hold K k0..k0+7 / k0+16..k0+23,
//              lanes 16-31 hold K k0+8..k0+15 / k0+24..k0+31.
__device__ __forceinline__ v16h load_a_f16(const _Float16* base, int ld, int k0) {
  const int l = lane_id(), r = l & 15, hi = l >> 4;
  const _Float16* p = base + (size_t)r * ld + k0 + hi * 8;
  v16h a;
  ((v8h*)&a)[0] = *(const v8h*)(p);
  ((v8h*)&a)[1] = *(const v8h*)(p + 16);
  return a;
}

// B 32x16 f16 (memory n-major with K contiguous): lane -> column (lane&15);
// lanes 0-15 K k0..k0+7 / k0+16..k0+23, lanes 16-31 K k0+8..k0+15 / k0+24..k0+31.
__device__ __forceinline__ v16h load_b_f16(const _Float16* base, int ld, int k0) {
  const int l = lane_id(), n = l & 15, hi = l >> 4;
  const _Float16* p = base + (size_t)n * ld + k0 + hi * 8;
  v16h b;
  ((v8h*)&b)[0] = *(const v8h*)(p);
  ((v8h*)&b)[1] = *(const v8h*)(p + 16);
  return b;
}

// ---------------------------------------------------------------------------
// Kernel 0: one-time fp32 -> f16 conversion (vector 4 per thread).
// ---------------------------------------------------------------------------
__global__ void __launch_bounds__(256)
f32_to_f16_kernel(const float* __restrict__ src, _Float16* __restrict__ dst, int n4) {
  const int i = blockIdx.x * 256 + threadIdx.x;
  if (i < n4) {
    v4f x = ((const v4f*)src)[i];
    v4h y;
#pragma unroll
    for (int j = 0; j < 4; ++j) y[j] = (_Float16)x[j];
    ((v4h*)dst)[i] = y;
  }
}

// ---------------------------------------------------------------------------
// Kernel 1: projection  out = X @ W^T + bias   (M=4096, N=1024, K=1024, f16 in)
// Per wave: 32x64 output tile. Per 32-K step: 12 b128 loads issued up front,
// then 8 WMMAs with staggered waits.
// out_mode 0: head-major f16  out[((b*H+h)*S + s)*dh + d]      (Q, K)
// out_mode 1: transposed f16  out[((b*H+h)*dh + d)*S + s]      (V)
// ---------------------------------------------------------------------------
__global__ void __launch_bounds__(256)
proj_qkv_kernel(const _Float16* __restrict__ X, const _Float16* __restrict__ W,
                const float* __restrict__ bias, _Float16* __restrict__ out,
                int out_mode) {
  const int wave  = threadIdx.x >> 5;
  const int task  = blockIdx.x * 8 + wave;   // 2048 wave tasks
  const int mtile = task >> 4;               // 0..127 (32 rows each)
  const int ng    = task & 15;               // 0..15 (64-column group)
  const int m0    = mtile * 32;
  const _Float16* A0 = X + (size_t)m0 * D_MODEL;
  const _Float16* A1 = X + (size_t)(m0 + 16) * D_MODEL;
  const _Float16* B0 = W + (size_t)(ng * 64 +  0) * D_MODEL;
  const _Float16* B1 = W + (size_t)(ng * 64 + 16) * D_MODEL;
  const _Float16* B2 = W + (size_t)(ng * 64 + 32) * D_MODEL;
  const _Float16* B3 = W + (size_t)(ng * 64 + 48) * D_MODEL;

  const v8f zero8 = {0.f, 0.f, 0.f, 0.f, 0.f, 0.f, 0.f, 0.f};
  v8f acc[2][4] = {{zero8, zero8, zero8, zero8}, {zero8, zero8, zero8, zero8}};

  for (int kk = 0; kk < D_MODEL; kk += 32) {
    // issue all fragment loads first (clauses), then the 8 WMMAs
    v16h a0 = load_a_f16(A0, D_MODEL, kk);
    v16h a1 = load_a_f16(A1, D_MODEL, kk);
    v16h b0 = load_b_f16(B0, D_MODEL, kk);
    v16h b1 = load_b_f16(B1, D_MODEL, kk);
    v16h b2 = load_b_f16(B2, D_MODEL, kk);
    v16h b3 = load_b_f16(B3, D_MODEL, kk);
    acc[0][0] = WMMA_F16(a0, b0, acc[0][0]);
    acc[1][0] = WMMA_F16(a1, b0, acc[1][0]);
    acc[0][1] = WMMA_F16(a0, b1, acc[0][1]);
    acc[1][1] = WMMA_F16(a1, b1, acc[1][1]);
    acc[0][2] = WMMA_F16(a0, b2, acc[0][2]);
    acc[1][2] = WMMA_F16(a1, b2, acc[1][2]);
    acc[0][3] = WMMA_F16(a0, b3, acc[0][3]);
    acc[1][3] = WMMA_F16(a1, b3, acc[1][3]);
    // unconditional speculative prefetch of the next A strip (dropped if OOB)
    __builtin_prefetch(A0 + (size_t)(lane_id() & 15) * D_MODEL + kk + 64, 0, 2);
  }

  const int l = lane_id(), ncl = l & 15, hi = l >> 4;
  if (out_mode == 0) {
#pragma unroll
    for (int mt = 0; mt < 2; ++mt)
#pragma unroll
      for (int t = 0; t < 4; ++t) {
        const int n = ng * 64 + t * 16 + ncl;
        const int h = n >> 6, d = n & 63;
        const float bv = bias[n];
#pragma unroll
        for (int r = 0; r < 8; ++r) {
          const int m = m0 + mt * 16 + r + hi * 8;
          const int bb = m >> 11, s = m & 2047;
          out[((size_t)((bb * HEADS + h) * SEQ + s)) * DH + d] =
              (_Float16)(acc[mt][t][r] + bv);
        }
      }
  } else {
#pragma unroll
    for (int mt = 0; mt < 2; ++mt)
#pragma unroll
      for (int t = 0; t < 4; ++t) {
        const int n = ng * 64 + t * 16 + ncl;
        const int h = n >> 6, d = n & 63;
        const float bv = bias[n];
#pragma unroll
        for (int r = 0; r < 8; ++r) {
          const int m = m0 + mt * 16 + r + hi * 8;
          const int bb = m >> 11, s = m & 2047;
          out[((size_t)((bb * HEADS + h) * DH + d)) * SEQ + s] =
              (_Float16)(acc[mt][t][r] + bv);
        }
      }
  }
}

// ---------------------------------------------------------------------------
// Kernel 2: attention core. One workgroup per (b,h, 32-row q block).
// 256KB dynamic LDS holds the 32x2048 score strip (f32).
//  phase 1: S = scale * Q Kᵀ via WMMA into LDS (2 key tiles in flight)
//  phase 2: row softmax (shfl reductions); write attn (f32, float2 stores) to
//           d_out ONCE; pack probabilities as f16 in-place into the low half
//           of each LDS row (safe: one wave owns a row; loads precede stores)
//  phase 3: ctx = P V via WMMA, even/odd split accumulators
// ---------------------------------------------------------------------------
__global__ void __launch_bounds__(256)
attn_kernel(const _Float16* __restrict__ qws, const _Float16* __restrict__ kws,
            const _Float16* __restrict__ vws, _Float16* __restrict__ ctx,
            float* __restrict__ attn_out) {
  extern __shared__ float sS[];              // 32 * 2048 floats = 256 KB
  const int bh = blockIdx.x >> 6;            // 0..31  (b*H + h)
  const int qb = blockIdx.x & 63;            // 0..63
  const int b  = bh >> 4, h = bh & 15;
  const int q0 = qb * 32;
  const _Float16* Qh = qws + (size_t)bh * SEQ * DH;
  const _Float16* Kh = kws + (size_t)bh * SEQ * DH;
  const _Float16* Vt = vws + (size_t)bh * DH * SEQ;
  const int wave = threadIdx.x >> 5;
  const int l = lane_id(), nl = l & 15, hi = l >> 4;
  const float scale = 0.125f;                // 1/sqrt(64)

  // ---- phase 1: scores (two key tiles per iteration, independent chains) ---
  for (int mt = 0; mt < 2; ++mt) {
    v16h a0 = load_a_f16(Qh + (size_t)(q0 + mt * 16) * DH, DH, 0);
    v16h a1 = load_a_f16(Qh + (size_t)(q0 + mt * 16) * DH, DH, 32);
    for (int i = 0; i < 16; i += 2) {
      const int nt0 = wave * 16 + i;         // 128 key tiles over 8 waves
      const int nt1 = nt0 + 1;
      v16h b00 = load_b_f16(Kh + (size_t)(nt0 * 16) * DH, DH, 0);
      v16h b01 = load_b_f16(Kh + (size_t)(nt0 * 16) * DH, DH, 32);
      v16h b10 = load_b_f16(Kh + (size_t)(nt1 * 16) * DH, DH, 0);
      v16h b11 = load_b_f16(Kh + (size_t)(nt1 * 16) * DH, DH, 32);
      v8f c0 = {0.f, 0.f, 0.f, 0.f, 0.f, 0.f, 0.f, 0.f};
      v8f c1 = {0.f, 0.f, 0.f, 0.f, 0.f, 0.f, 0.f, 0.f};
      c0 = WMMA_F16(a0, b00, c0);
      c1 = WMMA_F16(a0, b10, c1);
      c0 = WMMA_F16(a1, b01, c0);
      c1 = WMMA_F16(a1, b11, c1);
      const int row0 = mt * 16 + hi * 8;
#pragma unroll
      for (int r = 0; r < 8; ++r)
        sS[(size_t)(row0 + r) * SEQ + nt0 * 16 + nl] = c0[r] * scale;
#pragma unroll
      for (int r = 0; r < 8; ++r)
        sS[(size_t)(row0 + r) * SEQ + nt1 * 16 + nl] = c1[r] * scale;
    }
  }
  __syncthreads();

  // ---- phase 2: softmax; write attn once; pack f16 P into LDS ----
  for (int rr = 0; rr < 4; ++rr) {
    const int row = wave * 4 + rr;
    float* srow = sS + (size_t)row * SEQ;
    float mx = -3.0e38f;
    for (int i = l; i < SEQ; i += 32) mx = fmaxf(mx, srow[i]);
#pragma unroll
    for (int off = 16; off > 0; off >>= 1) mx = fmaxf(mx, __shfl_xor(mx, off, 32));
    float sum = 0.f;
    for (int i = l; i < SEQ; i += 32) {
      const float e = __expf(srow[i] - mx);
      srow[i] = e;
      sum += e;
    }
#pragma unroll
    for (int off = 16; off > 0; off >>= 1) sum += __shfl_xor(sum, off, 32);
    const float inv = 1.0f / sum;
    // torch layout: attn[(h*B + b), q, k]
    float* grow = attn_out + ((size_t)(h * BATCH + b) * SEQ + (q0 + row)) * SEQ;
    for (int i = 2 * l; i < SEQ; i += 64) {
      const float p0 = srow[i] * inv;
      const float p1 = srow[i + 1] * inv;
      float2 pr; pr.x = p0; pr.y = p1;
      *(float2*)(grow + i) = pr;             // coalesced 8B stores
      union { _Float16 hv[2]; unsigned int u; } pk;
      pk.hv[0] = (_Float16)p0;
      pk.hv[1] = (_Float16)p1;
      ((unsigned int*)srow)[i >> 1] = pk.u;  // f16 pack into low half of row
    }
  }
  __syncthreads();

  // ---- phase 3: context = P @ V  (8 waves -> 2 mtiles x 4 dh-tiles) ----
  {
    const int mt = wave >> 2;                // 0..1
    const int nt = wave & 3;                 // 0..3 (16-wide dh tile)
    // f16 P rows live at half-index stride 4096 (row = 8KB of LDS)
    const _Float16* Ph = (const _Float16*)sS + (size_t)(mt * 16) * 4096;
    const _Float16* Vb = Vt + (size_t)(nt * 16) * SEQ;
    v8f accE = {0.f, 0.f, 0.f, 0.f, 0.f, 0.f, 0.f, 0.f};
    v8f accO = {0.f, 0.f, 0.f, 0.f, 0.f, 0.f, 0.f, 0.f};
    for (int kk = 0; kk < SEQ; kk += 64) {
      v16h aE = load_a_f16(Ph, 4096, kk);
      v16h bE = load_b_f16(Vb, SEQ, kk);
      v16h aO = load_a_f16(Ph, 4096, kk + 32);
      v16h bO = load_b_f16(Vb, SEQ, kk + 32);
      accE = WMMA_F16(aE, bE, accE);
      accO = WMMA_F16(aO, bO, accO);
    }
    const v8f acc = accE + accO;
    const int dcol = nt * 16 + nl;
#pragma unroll
    for (int r = 0; r < 8; ++r) {
      const int q = q0 + mt * 16 + r + hi * 8;
      const size_t tok = (size_t)b * SEQ + q;
      ctx[tok * D_MODEL + h * DH + dcol] = (_Float16)acc[r];
    }
  }
}

// ---------------------------------------------------------------------------
// Kernel 3: output projection  ows = ctx @ Wo^T + bo + residual  (fp32 out)
// Per wave: 32x64 tile, all-f16 operands, loads hoisted before WMMAs.
// ---------------------------------------------------------------------------
__global__ void __launch_bounds__(256)
out_proj_kernel(const _Float16* __restrict__ ctx, const _Float16* __restrict__ Wo,
                const float* __restrict__ bo, const float* __restrict__ resid,
                float* __restrict__ ows) {
  const int wave  = threadIdx.x >> 5;
  const int task  = blockIdx.x * 8 + wave;
  const int mtile = task >> 4, ng = task & 15;
  const int m0    = mtile * 32;
  const _Float16* A0 = ctx + (size_t)m0 * D_MODEL;
  const _Float16* A1 = ctx + (size_t)(m0 + 16) * D_MODEL;
  const _Float16* B0 = Wo + (size_t)(ng * 64 +  0) * D_MODEL;
  const _Float16* B1 = Wo + (size_t)(ng * 64 + 16) * D_MODEL;
  const _Float16* B2 = Wo + (size_t)(ng * 64 + 32) * D_MODEL;
  const _Float16* B3 = Wo + (size_t)(ng * 64 + 48) * D_MODEL;

  const v8f zero8 = {0.f, 0.f, 0.f, 0.f, 0.f, 0.f, 0.f, 0.f};
  v8f acc[2][4] = {{zero8, zero8, zero8, zero8}, {zero8, zero8, zero8, zero8}};

  for (int kk = 0; kk < D_MODEL; kk += 32) {
    v16h a0 = load_a_f16(A0, D_MODEL, kk);
    v16h a1 = load_a_f16(A1, D_MODEL, kk);
    v16h b0 = load_b_f16(B0, D_MODEL, kk);
    v16h b1 = load_b_f16(B1, D_MODEL, kk);
    v16h b2 = load_b_f16(B2, D_MODEL, kk);
    v16h b3 = load_b_f16(B3, D_MODEL, kk);
    acc[0][0] = WMMA_F16(a0, b0, acc[0][0]);
    acc[1][0] = WMMA_F16(a1, b0, acc[1][0]);
    acc[0][1] = WMMA_F16(a0, b1, acc[0][1]);
    acc[1][1] = WMMA_F16(a1, b1, acc[1][1]);
    acc[0][2] = WMMA_F16(a0, b2, acc[0][2]);
    acc[1][2] = WMMA_F16(a1, b2, acc[1][2]);
    acc[0][3] = WMMA_F16(a0, b3, acc[0][3]);
    acc[1][3] = WMMA_F16(a1, b3, acc[1][3]);
    __builtin_prefetch(A0 + (size_t)(lane_id() & 15) * D_MODEL + kk + 64, 0, 2);
  }

  const int l = lane_id(), ncl = l & 15, hi = l >> 4;
#pragma unroll
  for (int mt = 0; mt < 2; ++mt)
#pragma unroll
    for (int t = 0; t < 4; ++t) {
      const int n = ng * 64 + t * 16 + ncl;
      const float bv = bo[n];
#pragma unroll
      for (int r = 0; r < 8; ++r) {
        const int m = m0 + mt * 16 + r + hi * 8;
        const size_t idx = (size_t)m * D_MODEL + n;
        ows[idx] = acc[mt][t][r] + bv + resid[idx];
      }
    }
}

// ---------------------------------------------------------------------------
// Kernel 4: row LayerNorm (D=1024), one 256-thread block per token row.
// ---------------------------------------------------------------------------
__global__ void __launch_bounds__(256)
layernorm_kernel(const float* __restrict__ ows, const float* __restrict__ gamma,
                 const float* __restrict__ beta, float* __restrict__ out) {
  const int row = blockIdx.x;
  const float* x = ows + (size_t)row * D_MODEL;
  float v[4];
  float s = 0.f, s2 = 0.f;
#pragma unroll
  for (int k = 0; k < 4; ++k) {
    const int i = (int)threadIdx.x + k * 256;
    v[k] = x[i];
    s += v[k];
    s2 += v[k] * v[k];
  }
#pragma unroll
  for (int off = 16; off > 0; off >>= 1) {
    s  += __shfl_xor(s, off, 32);
    s2 += __shfl_xor(s2, off, 32);
  }
  __shared__ float rs[8], rs2[8];
  const int wave = threadIdx.x >> 5;
  if (lane_id() == 0) { rs[wave] = s; rs2[wave] = s2; }
  __syncthreads();
  __shared__ float stat[2];
  if (wave == 0) {
    float a  = (lane_id() < 8) ? rs[lane_id()]  : 0.f;
    float a2 = (lane_id() < 8) ? rs2[lane_id()] : 0.f;
#pragma unroll
    for (int off = 4; off > 0; off >>= 1) {
      a  += __shfl_xor(a, off, 32);
      a2 += __shfl_xor(a2, off, 32);
    }
    if (lane_id() == 0) { stat[0] = a; stat[1] = a2; }
  }
  __syncthreads();
  const float mu  = stat[0] * (1.0f / (float)D_MODEL);
  const float var = stat[1] * (1.0f / (float)D_MODEL) - mu * mu;
  const float inv = rsqrtf(var + 1e-5f);
#pragma unroll
  for (int k = 0; k < 4; ++k) {
    const int i = (int)threadIdx.x + k * 256;
    out[(size_t)row * D_MODEL + i] = (v[k] - mu) * inv * gamma[i] + beta[i];
  }
}

// ---------------------------------------------------------------------------
extern "C" void kernel_launch(void* const* d_in, const int* in_sizes, int n_in,
                              void* d_out, int out_size, void* d_ws, size_t ws_size,
                              hipStream_t stream) {
  const float* query = (const float*)d_in[0];
  const float* key_  = (const float*)d_in[1];
  const float* value = (const float*)d_in[2];
  const float* Wq = (const float*)d_in[3];
  const float* bq = (const float*)d_in[4];
  const float* Wk = (const float*)d_in[5];
  const float* bk = (const float*)d_in[6];
  const float* Wv = (const float*)d_in[7];
  const float* bv = (const float*)d_in[8];
  const float* Wo = (const float*)d_in[9];
  const float* bo = (const float*)d_in[10];
  const float* gamma = (const float*)d_in[11];
  const float* beta_ = (const float*)d_in[12];

  float* outp  = (float*)d_out;                            // (B,S,D) fp32
  float* attnp = outp + (size_t)MTOK * D_MODEL;            // (H*B,S,S) fp32

  // workspace carve (~80 MB)
  const size_t NX = (size_t)MTOK * D_MODEL;                // 4 Mi elements
  const size_t NW = (size_t)D_MODEL * D_MODEL;             // 1 Mi elements
  _Float16* xqf = (_Float16*)d_ws;                         // f16 activations
  _Float16* xkf = xqf + NX;
  _Float16* xvf = xkf + NX;
  _Float16* wqf = xvf + NX;                                // f16 weights
  _Float16* wkf = wqf + NW;
  _Float16* wvf = wkf + NW;
  _Float16* wof = wvf + NW;
  _Float16* qws = wof + NW;                                // projected heads
  _Float16* kws = qws + NX;
  _Float16* vws = kws + NX;                                // transposed V
  _Float16* ctx = vws + NX;
  float*    ows = (float*)(ctx + NX);                      // fp32 pre-LN

  const dim3 blk(256);

  // one-time f16 conversion pass
  f32_to_f16_kernel<<<(int)(NX / 4 / 256), blk, 0, stream>>>(query, xqf, (int)(NX / 4));
  f32_to_f16_kernel<<<(int)(NX / 4 / 256), blk, 0, stream>>>(key_,  xkf, (int)(NX / 4));
  f32_to_f16_kernel<<<(int)(NX / 4 / 256), blk, 0, stream>>>(value, xvf, (int)(NX / 4));
  f32_to_f16_kernel<<<(int)(NW / 4 / 256), blk, 0, stream>>>(Wq, wqf, (int)(NW / 4));
  f32_to_f16_kernel<<<(int)(NW / 4 / 256), blk, 0, stream>>>(Wk, wkf, (int)(NW / 4));
  f32_to_f16_kernel<<<(int)(NW / 4 / 256), blk, 0, stream>>>(Wv, wvf, (int)(NW / 4));
  f32_to_f16_kernel<<<(int)(NW / 4 / 256), blk, 0, stream>>>(Wo, wof, (int)(NW / 4));

  proj_qkv_kernel<<<256, blk, 0, stream>>>(xqf, wqf, bq, qws, 0);
  proj_qkv_kernel<<<256, blk, 0, stream>>>(xkf, wkf, bk, kws, 0);
  proj_qkv_kernel<<<256, blk, 0, stream>>>(xvf, wvf, bv, vws, 1);

  (void)hipFuncSetAttribute((const void*)attn_kernel,
                            hipFuncAttributeMaxDynamicSharedMemorySize, 262144);
  attn_kernel<<<2048, blk, 262144, stream>>>(qws, kws, vws, ctx, attnp);

  out_proj_kernel<<<256, blk, 0, stream>>>(ctx, wof, bo, query, ows);
  layernorm_kernel<<<4096, blk, 0, stream>>>(ows, gamma, beta_, outp);
}
